// CrossAttnHead_74165495267769
// MI455X (gfx1250) — matrix operand order repaired and checked by reading, not compiled, not run on hardware
//
#include <hip/hip_runtime.h>

// ---------------------------------------------------------------------------
// CrossAttnHead for MI455X (gfx1250, wave32, WMMA + async global->LDS).
//
// Algebraic rewrite: out-proj + mean-pool + rank-1 classifier collapse into
//   wef[d]    = sum_e w_cls[e] * w_out[e,d]
//   vw[n,k,h] = kv[n,k,:] . (sum_{d in head h} wef[d]*Wv[d,:]) + bv_eff[h]
//   scores[n,b] = ( (1/T) * sum_{h,t} softmax_k(q.k^T/sqrt(dh)) . vw[n,:,h]
//                   + (w_cls.b_out + b_cls) ) / tau
// so attention*V is a softmax-weighted scalar average and the dominant work
// is Q.K^T (18.4 GFLOP) -> v_wmma_f32_16x16x32_bf16 with f32 softmax.
//
// Data movement: the 16x384 bf16 K-tile (12 KB) consumed by all 8 heads of a
// block is staged ONCE into LDS, double-buffered, prefetched with
// GLOBAL_LOAD_ASYNC_TO_LDS_B128 (ASYNCcnt / s_wait_asynccnt), overlapping
// global latency with WMMA + softmax.
//
// Workspace (~7.7 MiB of d_ws, float offsets):
//   acc[125] | scal[8] | bv_eff[8] | wv_eff[8*384] | rowmap[4900]i32 |
//   Qb bf16[25*208*384] | Kb bf16[5*992*384] | VW f32[4900*8]
// ---------------------------------------------------------------------------

typedef __attribute__((ext_vector_type(16))) __bf16 v16bf;
typedef __attribute__((ext_vector_type(8)))  float  v8f;
typedef __attribute__((ext_vector_type(4)))  int    v4i;
typedef unsigned short u16;
typedef unsigned int   u32;

#define DM   384
#define TT   196
#define BQN  25
#define NWAY 5
#define HN   8
#define DH   48
#define QPAD 208          // 196 padded to 16
#define KROWS 980         // 5*196 keys per class
#define KPAD 992          // padded to 16
#define QTILES 13         // 208/16
#define KTILES 62         // 992/16

#if defined(__AMDGCN__) && \
    __has_builtin(__builtin_amdgcn_global_load_async_to_lds_b128) && \
    __has_builtin(__builtin_amdgcn_s_wait_asynccnt)
#define HAVE_ASYNC_LDS 1
typedef __attribute__((address_space(1))) v4i* as1_v4i_p;   // global b128 payload
typedef __attribute__((address_space(3))) v4i* as3_v4i_p;   // LDS b128 payload
#else
#define HAVE_ASYNC_LDS 0
#endif

__device__ __forceinline__ u16 f2bf_us(float f) {
  u32 u = __builtin_bit_cast(u32, f);
  u32 r = u + 0x7fffu + ((u >> 16) & 1u);      // round-to-nearest-even
  return (u16)(r >> 16);
}
__device__ __forceinline__ __bf16 us2bf(u16 s) { return __builtin_bit_cast(__bf16, s); }

union Frag16 { v16bf v; uint4 q[2]; };

// A-matrix 16x32 bf16 layout (ISA 7.12.2): lane m=lane&15, kb=(lane>>4)*8,
// elements k = [kb..kb+7] then [kb+16..kb+23], each an aligned 16B chunk.
__device__ __forceinline__ v16bf frag_a_bf16(const u16* row, int kb) {
  Frag16 f;
  f.q[0] = *reinterpret_cast<const uint4*>(row + kb);
  f.q[1] = *reinterpret_cast<const uint4*>(row + kb + 16);
  return f.v;
}
__device__ __forceinline__ v16bf frag_a_half(const u16* row, int kb) { // upper 16 k's = 0
  Frag16 f;
  f.q[0] = *reinterpret_cast<const uint4*>(row + kb);
  f.q[1] = uint4{0u, 0u, 0u, 0u};
  return f.v;
}
// B operand, contiguous k order used here (A matches): lane n=lane&15 holds K=[ks..ks+15], ks=(lane>>4)*16,
// contiguous in our row-major storage (we store B^T rows).
__device__ __forceinline__ v16bf frag_b_contig(const u16* p) {
  Frag16 f;
  f.q[0] = *reinterpret_cast<const uint4*>(p);
  f.q[1] = *reinterpret_cast<const uint4*>(p + 8);
  return f.v;
}
__device__ __forceinline__ v16bf frag_zero() {
  Frag16 f;
  f.q[0] = uint4{0u, 0u, 0u, 0u};
  f.q[1] = uint4{0u, 0u, 0u, 0u};
  return f.v;
}
__device__ __forceinline__ v16bf frag_b_f32(const float* p) { // 16 contiguous f32 -> bf16
  v16bf r;
#pragma unroll
  for (int i = 0; i < 16; ++i) r[i] = us2bf(f2bf_us(p[i]));
  return r;
}

__device__ __forceinline__ v8f wmma_bf16(v16bf a, v16bf b, v8f c) {
  return __builtin_amdgcn_wmma_f32_16x16x32_bf16(false, a, false, b, (short)0, c,
                                                 false, false);
}

// Stage one 16x384 bf16 K-tile (12 KB, contiguous in Kb) into LDS.
// 768 x 16B chunks; 256 threads x 3 transfers.
__device__ __forceinline__ void stage_k_tile(const u16* __restrict__ g, u16* l) {
  const int t = threadIdx.x;
#pragma unroll
  for (int i = 0; i < 3; ++i) {
    const int off = (t + i * 256) * 8;                 // u16 units, 8 u16 = 16B
#if HAVE_ASYNC_LDS
    __builtin_amdgcn_global_load_async_to_lds_b128(
        (as1_v4i_p)(g + off), (as3_v4i_p)(l + off), 0, 0);
#else
    *reinterpret_cast<uint4*>(l + off) = *reinterpret_cast<const uint4*>(g + off);
#endif
  }
}
__device__ __forceinline__ void wait_async_all() {
#if HAVE_ASYNC_LDS
  __builtin_amdgcn_s_wait_asynccnt(0);
#endif
}

// ---------------------------------------------------------------------------
// Kernel 0: order/rowmap, wef, wv_eff/bv_eff, constants, zero acc. 1 block, 384 thr.
// ---------------------------------------------------------------------------
__global__ void prep_kernel(const long long* __restrict__ label_support,
                            const int* __restrict__ n_way_p,
                            const float* __restrict__ w_cls,
                            const float* __restrict__ w_out,
                            const float* __restrict__ b_out,
                            const float* __restrict__ b_cls,
                            const float* __restrict__ b_in,
                            const float* __restrict__ w_in,
                            const float* __restrict__ log_tau,
                            float* __restrict__ acc, float* __restrict__ scal,
                            float* __restrict__ bv_eff, float* __restrict__ wv_eff,
                            int* __restrict__ rowmap) {
  __shared__ int   s_order[32];
  __shared__ float s_wef[DM];
  const int tid = threadIdx.x;
  int nway = n_way_p[0];                 // int32 view works for int32/int64 LE
  if (nway <= 0 || nway > BQN) nway = NWAY;
  const int kshot = BQN / nway;

  if (tid == 0) {                        // stable counting sort by label
    int idx = 0;
    for (int c = 0; c < nway; ++c)
      for (int i = 0; i < BQN; ++i)
        if ((int)label_support[i] == c) s_order[idx++] = i;
    float ct = b_cls[0];
    for (int e = 0; e < DM; ++e) ct += w_cls[e] * b_out[e];
    scal[0] = ct;
    scal[1] = 1.f / (__expf(log_tau[0]) + 1e-8f);
  }
  if (tid < 125) acc[tid] = 0.f;
  if (tid < DM) {
    float s = 0.f;
    for (int e = 0; e < DM; ++e) s += w_cls[e] * w_out[e * DM + tid];
    s_wef[tid] = s;
  }
  __syncthreads();

  for (int i = tid; i < NWAY * KROWS; i += blockDim.x) {
    int n = i / KROWS, j = i - n * KROWS;
    int sic = j / TT, t = j - sic * TT;
    rowmap[i] = s_order[n * kshot + sic] * TT + t;
  }
  for (int i = tid; i < HN * DM; i += blockDim.x) {
    int h = i / DM, c = i - h * DM;
    float s = 0.f;
    for (int dd = 0; dd < DH; ++dd)
      s += s_wef[h * DH + dd] * w_in[(2 * DM + h * DH + dd) * DM + c];
    wv_eff[i] = s;
  }
  if (tid < HN) {
    float s = 0.f;
    for (int dd = 0; dd < DH; ++dd)
      s += s_wef[tid * DH + dd] * b_in[2 * DM + tid * DH + dd];
    bv_eff[tid] = s;
  }
}

// ---------------------------------------------------------------------------
// Kernel 1: Q/K projections -> bf16, via WMMA.  grid=(635,3), block=256 (8 waves).
// tiles 0..324: Q (25*208 padded rows); 325..634: K (5*992 padded rows).
// ---------------------------------------------------------------------------
__global__ void proj_kernel(const float* __restrict__ emb_support,
                            const float* __restrict__ emb_query,
                            const float* __restrict__ w_in,
                            const float* __restrict__ b_in,
                            const int* __restrict__ rowmap,
                            u16* __restrict__ Qb, u16* __restrict__ Kb) {
  __shared__ __align__(16) u16 sA[16 * 32];
  const int tile = blockIdx.x;
  const int wave = threadIdx.x >> 5;
  const int lane = threadIdx.x & 31;
  const int ncol = blockIdx.y * 128 + wave * 16;
  const bool isQ = (tile < 325);
  const int  t2  = tile - 325;

  // staging rows owned by this thread: r0 = wave, r1 = wave+8, column = lane
  const float* srcp[2];
#pragma unroll
  for (int half = 0; half < 2; ++half) {
    int r = wave + half * 8;
    const float* p = nullptr;
    if (isQ) {
      int gr = tile * 16 + r, bq = gr / QPAD, qp = gr - bq * QPAD;
      if (qp < TT) p = emb_query + (bq * TT + qp) * DM;
    } else {
      int n = t2 / KTILES, kp = (t2 - n * KTILES) * 16 + r;
      if (kp < KROWS) p = emb_support + rowmap[n * KROWS + kp] * DM;
    }
    srcp[half] = p;
  }
  const int woff = isQ ? 0 : DM;                        // Wq vs Wk rows of w_in
  const float* wrow = w_in + (woff + ncol + (lane & 15)) * DM + ((lane >> 4) * 16);

  v8f c = {};
  const int kb = (lane >> 4) * 8;
  for (int kc = 0; kc < DM; kc += 32) {
    sA[wave * 32 + lane]       = srcp[0] ? f2bf_us(srcp[0][kc + lane]) : (u16)0;
    sA[(wave + 8) * 32 + lane] = srcp[1] ? f2bf_us(srcp[1][kc + lane]) : (u16)0;
    __syncthreads();
    v16bf a  = frag_b_contig(sA + (lane & 15) * 32 + kb * 2);
    v16bf bf = frag_b_f32(wrow + kc);
    c = wmma_bf16(a, bf, c);
    __syncthreads();
  }

  const int nn = lane & 15;
  const float bias = b_in[woff + ncol + nn];
  u16* outbase = isQ ? (Qb + (tile * 16) * DM) : (Kb + (t2 * 16) * DM);
#pragma unroll
  for (int j = 0; j < 8; ++j) {
    int r = j + ((lane >> 4) << 3);                      // D-layout: M = j (+8 for hi half)
    outbase[r * DM + ncol + nn] = f2bf_us(c[j] + bias);
  }
}

// ---------------------------------------------------------------------------
// Kernel 2: per-key scalar values vw[n,k,h] (tiny, VALU).
// ---------------------------------------------------------------------------
__global__ void vw_kernel(const float* __restrict__ emb_support,
                          const float* __restrict__ wv_eff,
                          const float* __restrict__ bv_eff,
                          const int* __restrict__ rowmap,
                          float* __restrict__ VW) {
  int i = blockIdx.x * blockDim.x + threadIdx.x;
  if (i >= NWAY * KROWS * HN) return;
  int r = i >> 3, h = i & 7;
  const float* x = emb_support + rowmap[r] * DM;
  const float* w = wv_eff + h * DM;
  float s = bv_eff[h];
  for (int c = 0; c < DM; ++c) s += x[c] * w[c];
  VW[i] = s;
}

// ---------------------------------------------------------------------------
// Kernel 3: fused attention + pooled rank-1 epilogue.
// grid=(125, 13), block=256: block=(n,b), y=q-tile, wave=head.
// K-tiles (16x384 bf16 = 12KB) staged once per block into LDS, double-buffered
// with async global->LDS prefetch; all 8 heads read their 48-col slice via ds.
// ---------------------------------------------------------------------------
__global__ void attn_kernel(const u16* __restrict__ Qb, const u16* __restrict__ Kb,
                            const float* __restrict__ VW, float* __restrict__ acc) {
  __shared__ __align__(16) u16 sK[2][16 * DM];          // 2 x 12 KB
  const int n = blockIdx.x / BQN, b = blockIdx.x - n * BQN;
  const int qt = blockIdx.y;
  const int h = threadIdx.x >> 5;
  const int lane = threadIdx.x & 31;
  const int mn = lane & 15;            // A-row / B-col / D-col for this lane
  const int kb = (lane >> 4) * 8;
  const int ks = (lane >> 4) * 16;
  const float scale = 0.14433756729740643f;   // 1/sqrt(48)

  const u16* qrow = Qb + (b * QPAD + qt * 16 + mn) * DM + h * DH;
  v16bf a0 = frag_b_contig(qrow + ks);        // d = 0..31
  v16bf a1 = (lane < 16) ? frag_b_contig(qrow + 32) : frag_zero(); // d = 32..47

  float mr[8], sr[8], wr[8];
#pragma unroll
  for (int j = 0; j < 8; ++j) { mr[j] = -1e30f; sr[j] = 0.f; wr[j] = 0.f; }

  const u16* kgbase = Kb + (n * KPAD) * DM;   // 16 rows per tile, contiguous
  stage_k_tile(kgbase, &sK[0][0]);            // prefetch tile 0

  for (int kt = 0; kt < KTILES; ++kt) {
    wait_async_all();                         // my copies for buf[kt&1] landed
    __syncthreads();                          // everyone's copies landed; prev reads done
    if (kt + 1 < KTILES)                      // prefetch next tile into other buffer
      stage_k_tile(kgbase + (kt + 1) * 16 * DM, &sK[(kt + 1) & 1][0]);

    const u16* kl = &sK[kt & 1][0] + mn * DM + h * DH;
    v16bf b0 = frag_b_contig(kl + ks);                             // d = 0..31
    v16bf b1 = (lane < 16) ? frag_b_contig(kl + 32) : frag_zero(); // d = 32..47
    v8f c = {};
    c = wmma_bf16(a0, b0, c);
    c = wmma_bf16(a1, b1, c);

    const int gk = kt * 16 + mn;
    const bool kvalid = gk < KROWS;
    const float vwv = kvalid ? VW[(n * KROWS + gk) * HN + h] : 0.f;
#pragma unroll
    for (int j = 0; j < 8; ++j) {
      float val = kvalid ? c[j] * scale : -1e30f;
      float tm = val;
      tm = fmaxf(tm, __shfl_xor(tm, 1));
      tm = fmaxf(tm, __shfl_xor(tm, 2));
      tm = fmaxf(tm, __shfl_xor(tm, 4));
      tm = fmaxf(tm, __shfl_xor(tm, 8));
      float nm = fmaxf(mr[j], tm);
      float corr = __expf(mr[j] - nm);
      float p  = __expf(val - nm);
      float pv = p * vwv;
      p += __shfl_xor(p, 1);  pv += __shfl_xor(pv, 1);
      p += __shfl_xor(p, 2);  pv += __shfl_xor(pv, 2);
      p += __shfl_xor(p, 4);  pv += __shfl_xor(pv, 4);
      p += __shfl_xor(p, 8);  pv += __shfl_xor(pv, 8);
      sr[j] = sr[j] * corr + p;
      wr[j] = wr[j] * corr + pv;
      mr[j] = nm;
    }
  }

  float t = 0.f;
#pragma unroll
  for (int j = 0; j < 8; ++j) {
    int q = qt * 16 + j + ((lane >> 4) << 3);
    if (q < TT) t += wr[j] / sr[j];
  }
  t += __shfl_xor(t, 16);              // combine the two row-halves
  if (lane == 0) atomicAdd(acc + blockIdx.x, t);
}

// ---------------------------------------------------------------------------
// Kernel 4: finalize -> (25,5) fp32 output, transposed.
// ---------------------------------------------------------------------------
__global__ void fin_kernel(const float* __restrict__ acc,
                           const float* __restrict__ scal,
                           float* __restrict__ out) {
  int i = threadIdx.x;
  if (i < NWAY * BQN) {
    int n = i / BQN, b = i - n * BQN;
    out[b * NWAY + n] = (acc[i] * (1.f / (float)TT) + scal[0]) * scal[1];
  }
}

// ---------------------------------------------------------------------------
extern "C" void kernel_launch(void* const* d_in, const int* in_sizes, int n_in,
                              void* d_out, int out_size, void* d_ws, size_t ws_size,
                              hipStream_t stream) {
  const float*     emb_support   = (const float*)d_in[0];
  const float*     emb_query     = (const float*)d_in[1];
  const float*     w_in          = (const float*)d_in[2];
  const float*     b_in          = (const float*)d_in[3];
  const float*     w_out         = (const float*)d_in[4];
  const float*     b_out         = (const float*)d_in[5];
  const float*     w_cls         = (const float*)d_in[6];
  const float*     b_cls         = (const float*)d_in[7];
  const float*     log_tau       = (const float*)d_in[8];
  const long long* label_support = (const long long*)d_in[9];
  const int*       n_way         = (const int*)d_in[10];
  float* out = (float*)d_out;

  // workspace carve-up (float offsets; total ~7.7 MiB)
  float* ws     = (float*)d_ws;
  float* acc    = ws;                       // 125 (use 128)
  float* scal   = ws + 128;                 // 8
  float* bv_eff = ws + 136;                 // 8
  float* wv_eff = ws + 144;                 // 8*384 = 3072
  int*   rowmap = (int*)(ws + 3216);        // 4900 (use 4912)
  u16*   Qb     = (u16*)(ws + 8128);        // 25*208*384 bf16
  u16*   Kb     = Qb + BQN * QPAD * DM;     // 5*992*384 bf16
  float* VW     = (float*)(ws + 8128 + (BQN * QPAD * DM + NWAY * KPAD * DM) / 2);

  prep_kernel<<<1, 384, 0, stream>>>(label_support, n_way, w_cls, w_out, b_out,
                                     b_cls, b_in, w_in, log_tau,
                                     acc, scal, bv_eff, wv_eff, rowmap);
  proj_kernel<<<dim3(635, 3), 256, 0, stream>>>(emb_support, emb_query, w_in, b_in,
                                                rowmap, Qb, Kb);
  vw_kernel<<<(NWAY * KROWS * HN + 255) / 256, 256, 0, stream>>>(
      emb_support, wv_eff, bv_eff, rowmap, VW);
  attn_kernel<<<dim3(NWAY * BQN, QTILES), 256, 0, stream>>>(Qb, Kb, VW, acc);
  fin_kernel<<<1, 128, 0, stream>>>(acc, scal, out);
}
